// GroupedQueryAttention_30906584662338
// MI455X (gfx1250) — matrix operand order repaired and checked by reading, not compiled, not run on hardware
//
#include <hip/hip_runtime.h>
#include <hip/hip_bf16.h>

// ---------------------------------------------------------------------------
// GQA attention for MI455X (gfx1250), all matmuls through v_wmma_f32_16x16x32_f16
// Multi-tile GEMM waves (one A fragment amortized over 3-6 WMMAs).
// ---------------------------------------------------------------------------

typedef __attribute__((ext_vector_type(16))) _Float16 v16h;
typedef __attribute__((ext_vector_type(8)))  _Float16 v8h;
typedef __attribute__((ext_vector_type(8)))  float    v8f;

namespace cfg {
constexpr int HQ = 8;
constexpr int HKV = 2;
constexpr int DH = 48;      // head dim
constexpr int DP = 64;      // head dim padded to 2 x K=32 WMMA steps
constexpr int NE = 384;
constexpr int GROUP = 4;
constexpr int B = 8;
constexpr int T = 2048;
constexpr int M = B * T;    // 16384 rows
}

// ---------------------------------------------------------------------------
// WMMA fragment loaders (wave32, 16x16x32 f16)
//
// A (16xK row-major): lane l -> row m = l&15 ;
//   16 contiguous K halves starting at (l>>4)*16.
// B (Kx16, source stored N-major i.e. BT[n][k]): lane l -> col n = l&15 ;
//   16 contiguous K halves starting at (l>>4)*16.
// C/D (16x16 f32): VGPR r, lane l -> row = r + 8*(l>>4), col = l&15.
// ---------------------------------------------------------------------------

__device__ __forceinline__ v16h load_frag_a(const _Float16* base, int ld,
                                            int row0, int k0) {
  const int l  = threadIdx.x & 31;
  const int m  = l & 15;
  const int ko = (l >> 4) * 16;
  const _Float16* p = base + (size_t)(row0 + m) * ld + k0 + ko;
  return *(const v16h*)p;
}

__device__ __forceinline__ v16h load_frag_b(const _Float16* baseT, int ld,
                                            int col0, int k0) {
  const int l  = threadIdx.x & 31;
  const int n  = l & 15;
  const int ko = (l >> 4) * 16;
  const _Float16* p = baseT + (size_t)(col0 + n) * ld + k0 + ko;
  return *(const v16h*)p;
}

// ---------------------------------------------------------------------------
// Prep kernels
// ---------------------------------------------------------------------------

__global__ __launch_bounds__(256) void convert_f32_to_f16(const float* __restrict__ src,
                                                          _Float16* __restrict__ dst,
                                                          size_t n) {
  size_t i = (size_t)blockIdx.x * blockDim.x + threadIdx.x;
  if (i < n) dst[i] = (_Float16)src[i];
}

// dst[n*K + k] = (f16) src[k*N + n]   (weight K x N  ->  N-major f16)
__global__ __launch_bounds__(256) void transpose_to_f16(const float* __restrict__ src,
                                                        _Float16* __restrict__ dst,
                                                        int K, int N) {
  size_t i = (size_t)blockIdx.x * blockDim.x + threadIdx.x;
  if (i >= (size_t)K * N) return;
  int k = (int)(i % K);
  int n = (int)(i / K);
  dst[(size_t)n * K + k] = (_Float16)src[(size_t)k * N + n];
}

// ---------------------------------------------------------------------------
// Multi-tile WMMA GEMM: each wave computes a 16 x (16*NT) strip.
// One A fragment per K-step feeds NT WMMAs (A traffic cut NT-fold).
// C(f32) = A(f16,row-major) * BT(f16,N-major) + bias
// ---------------------------------------------------------------------------

template <int NT>
__global__ __launch_bounds__(32) void gemm_wmma_multi(const _Float16* __restrict__ A, int lda,
                                                      const _Float16* __restrict__ BT, int ldb,
                                                      const float* __restrict__ bias,
                                                      float* __restrict__ C, int ldc,
                                                      int n_groups, int K) {
  const int tile = blockIdx.x;
  const int tm = tile / n_groups;
  const int tg = tile % n_groups;
  const int row0 = tm * 16;
  const int col0 = tg * 16 * NT;

  v8f acc[NT];
#pragma unroll
  for (int nt = 0; nt < NT; ++nt) acc[nt] = (v8f){};

  for (int k0 = 0; k0 < K; k0 += 32) {
    if (k0 + 32 < K) {
      __builtin_prefetch((const void*)(A + (size_t)row0 * lda + k0 + 32), 0, 1);
      __builtin_prefetch((const void*)(BT + (size_t)col0 * ldb + k0 + 32), 0, 1);
    }
    v16h a = load_frag_a(A, lda, row0, k0);
#pragma unroll
    for (int nt = 0; nt < NT; ++nt) {
      v16h b = load_frag_b(BT, ldb, col0 + 16 * nt, k0);
      acc[nt] = __builtin_amdgcn_wmma_f32_16x16x32_f16(false, a, false, b,
                                                       (short)0, acc[nt], false, false);
    }
  }

  const int l = threadIdx.x & 31;
  const int n = l & 15;
  const int rb = (l >> 4) * 8;
#pragma unroll
  for (int nt = 0; nt < NT; ++nt) {
    const float bv = bias ? bias[col0 + 16 * nt + n] : 0.0f;
#pragma unroll
    for (int r = 0; r < 8; ++r) {
      C[(size_t)(row0 + rb + r) * ldc + col0 + 16 * nt + n] = acc[nt][r] + bv;
    }
  }
}

// V projection: full 96-wide strip per wave (NT=6), scattered to
// vt[b][h][d][t] (f16, t contiguous) for contiguous PV B-fragments.
__global__ __launch_bounds__(32) void gemm_wmma_vproj(const _Float16* __restrict__ A, int lda,
                                                      const _Float16* __restrict__ BT, int ldb,
                                                      _Float16* __restrict__ vt,
                                                      int K) {
  constexpr int NT = (cfg::HKV * cfg::DH) / 16;  // 6
  const int row0 = blockIdx.x * 16;

  v8f acc[NT];
#pragma unroll
  for (int nt = 0; nt < NT; ++nt) acc[nt] = (v8f){};

  for (int k0 = 0; k0 < K; k0 += 32) {
    v16h a = load_frag_a(A, lda, row0, k0);
#pragma unroll
    for (int nt = 0; nt < NT; ++nt) {
      v16h b = load_frag_b(BT, ldb, 16 * nt, k0);
      acc[nt] = __builtin_amdgcn_wmma_f32_16x16x32_f16(false, a, false, b,
                                                       (short)0, acc[nt], false, false);
    }
  }

  const int l = threadIdx.x & 31;
  const int n = l & 15;
  const int rb = (l >> 4) * 8;
#pragma unroll
  for (int nt = 0; nt < NT; ++nt) {
    const int nn = 16 * nt + n;
    const int h = nn / cfg::DH;
    const int d = nn % cfg::DH;
#pragma unroll
    for (int r = 0; r < 8; ++r) {
      int m = row0 + rb + r;
      int b_ = m / cfg::T;
      int t_ = m % cfg::T;
      vt[(((size_t)(b_ * cfg::HKV + h) * cfg::DH + d) * cfg::T) + t_] = (_Float16)acc[nt][r];
    }
  }
}

// ---------------------------------------------------------------------------
// RoPE + pack: src [B*T][nheads*48] f32 -> dst [B][nheads][T][64] f16 (zero pad)
// One thread per (b,t,h,pair i in [0,32)).
// ---------------------------------------------------------------------------

__global__ __launch_bounds__(256) void rope_pack(const float* __restrict__ src,
                                                 _Float16* __restrict__ dst,
                                                 int nheads) {
  size_t idx = (size_t)blockIdx.x * blockDim.x + threadIdx.x;
  size_t total = (size_t)cfg::B * cfg::T * nheads * 32;
  if (idx >= total) return;
  int i = (int)(idx % 32);
  int h = (int)((idx / 32) % nheads);
  int t = (int)((idx / (32 * (size_t)nheads)) % cfg::T);
  int b = (int)(idx / (32 * (size_t)nheads * cfg::T));

  _Float16* drow = dst + ((size_t)(b * nheads + h) * cfg::T + t) * cfg::DP;
  if (i < cfg::DH / 2) {
    const float* srow = src + ((size_t)b * cfg::T + t) * (nheads * cfg::DH) + h * cfg::DH;
    float xe = srow[2 * i];
    float xo = srow[2 * i + 1];
    float freq = __powf(10000.0f, -(float)(2 * i) / (float)cfg::DH);
    float ang = (float)t * freq;
    float s, c;
    __sincosf(ang, &s, &c);
    drow[2 * i]     = (_Float16)(xe * c - xo * s);
    drow[2 * i + 1] = (_Float16)(xe * s + xo * c);
  } else {
    drow[2 * i]     = (_Float16)0.0f;
    drow[2 * i + 1] = (_Float16)0.0f;
  }
}

// ---------------------------------------------------------------------------
// Flash attention: one wave per (b, hq, 16-row q tile).
// 32-key blocks: 4 WMMAs for scores (2 subtiles x 2 K-steps), online softmax
// in f32 VALU with shfl_xor row reductions, P bounced C->A layout via LDS,
// 3 WMMAs for PV (d tiles 0..2). Next block's K/V prefetched during softmax.
// ---------------------------------------------------------------------------

__global__ __launch_bounds__(32) void attn_flash(const _Float16* __restrict__ qh,
                                                 const _Float16* __restrict__ kh,
                                                 const _Float16* __restrict__ vt,
                                                 _Float16* __restrict__ ah) {
  __shared__ _Float16 plds[16 * 32];

  const int qtiles = cfg::T / 16;
  const int wid = blockIdx.x;
  const int qt = wid % qtiles;
  const int bh = wid / qtiles;
  const int h = bh % cfg::HQ;
  const int b = bh / cfg::HQ;
  const int hk = h / cfg::GROUP;

  const _Float16* qbase = qh + (size_t)(b * cfg::HQ + h) * cfg::T * cfg::DP;
  const _Float16* kbase = kh + (size_t)(b * cfg::HKV + hk) * cfg::T * cfg::DP;
  const _Float16* vbase = vt + (size_t)(b * cfg::HKV + hk) * cfg::DH * cfg::T;

  const v16h qa0 = load_frag_a(qbase, cfg::DP, qt * 16, 0);
  const v16h qa1 = load_frag_a(qbase, cfg::DP, qt * 16, 32);

  v8f o0 = {}, o1 = {}, o2 = {};
  float mrow[8], lrow[8];
#pragma unroll
  for (int r = 0; r < 8; ++r) { mrow[r] = -3.0e38f; lrow[r] = 0.0f; }

  const int l = threadIdx.x & 31;
  const int lane_col = l & 15;
  const int rhi = (l >> 4) * 8;
  const int tmax = qt * 16 + 15;
  const float scale = 0.14433756729740643f;  // 1/sqrt(48)

  for (int s0 = 0; s0 <= tmax; s0 += 32) {
    // ---- scores: 16 q rows x 32 keys, two 16x16 C tiles ----
    v8f sc0, sc1;
    {
      v16h kb0 = load_frag_b(kbase, cfg::DP, s0, 0);
      v16h kb1 = load_frag_b(kbase, cfg::DP, s0, 32);
      v8f c = {};
      c = __builtin_amdgcn_wmma_f32_16x16x32_f16(false, qa0, false, kb0, (short)0, c, false, false);
      c = __builtin_amdgcn_wmma_f32_16x16x32_f16(false, qa1, false, kb1, (short)0, c, false, false);
      sc0 = c;
    }
    {
      v16h kb0 = load_frag_b(kbase, cfg::DP, s0 + 16, 0);
      v16h kb1 = load_frag_b(kbase, cfg::DP, s0 + 16, 32);
      v8f c = {};
      c = __builtin_amdgcn_wmma_f32_16x16x32_f16(false, qa0, false, kb0, (short)0, c, false, false);
      c = __builtin_amdgcn_wmma_f32_16x16x32_f16(false, qa1, false, kb1, (short)0, c, false, false);
      sc1 = c;
    }

    // ---- overlap: prefetch next key block's K rows and V stripe while the
    //      softmax VALU chain below executes ----
    if (s0 + 32 <= tmax) {
      __builtin_prefetch((const void*)(kbase + (size_t)(s0 + 32 + l) * cfg::DP), 0, 1);
      __builtin_prefetch((const void*)(vbase + (size_t)l * cfg::T + s0 + 32), 0, 1);
    }

    // ---- scale + causal mask + new row max ----
    float mnew[8];
    const int sA = s0 + lane_col;
    const int sB = s0 + 16 + lane_col;
#pragma unroll
    for (int r = 0; r < 8; ++r) {
      const int trow = qt * 16 + rhi + r;
      float a = sc0[r] * scale;
      float bb = sc1[r] * scale;
      if (sA > trow) a = -1.0e30f;
      if (sB > trow) bb = -1.0e30f;
      sc0[r] = a;
      sc1[r] = bb;
      float mx = fmaxf(a, bb);
      mx = fmaxf(mx, __shfl_xor(mx, 1, 32));
      mx = fmaxf(mx, __shfl_xor(mx, 2, 32));
      mx = fmaxf(mx, __shfl_xor(mx, 4, 32));
      mx = fmaxf(mx, __shfl_xor(mx, 8, 32));
      mnew[r] = fmaxf(mrow[r], mx);
    }

    // ---- exp, row sum, rescale running state + accumulators ----
#pragma unroll
    for (int r = 0; r < 8; ++r) {
      float p0 = __expf(sc0[r] - mnew[r]);
      float p1 = __expf(sc1[r] - mnew[r]);
      sc0[r] = p0;
      sc1[r] = p1;
      float rs = p0 + p1;
      rs += __shfl_xor(rs, 1, 32);
      rs += __shfl_xor(rs, 2, 32);
      rs += __shfl_xor(rs, 4, 32);
      rs += __shfl_xor(rs, 8, 32);
      float corr = __expf(mrow[r] - mnew[r]);
      lrow[r] = lrow[r] * corr + rs;
      mrow[r] = mnew[r];
      o0[r] *= corr;
      o1[r] *= corr;
      o2[r] *= corr;
    }

    // ---- P: C-layout -> row-major f16 in LDS -> A fragment ----
#pragma unroll
    for (int r = 0; r < 8; ++r) {
      plds[(rhi + r) * 32 + lane_col]      = (_Float16)sc0[r];
      plds[(rhi + r) * 32 + 16 + lane_col] = (_Float16)sc1[r];
    }
    __syncthreads();
    v16h pa = load_frag_a(plds, 32, 0, 0);
    __syncthreads();

    // ---- PV: out += P (16x32) * V (32x48), three d tiles ----
    {
      v16h vb = load_frag_b(vbase, cfg::T, 0, s0);
      o0 = __builtin_amdgcn_wmma_f32_16x16x32_f16(false, pa, false, vb, (short)0, o0, false, false);
    }
    {
      v16h vb = load_frag_b(vbase, cfg::T, 16, s0);
      o1 = __builtin_amdgcn_wmma_f32_16x16x32_f16(false, pa, false, vb, (short)0, o1, false, false);
    }
    {
      v16h vb = load_frag_b(vbase, cfg::T, 32, s0);
      o2 = __builtin_amdgcn_wmma_f32_16x16x32_f16(false, pa, false, vb, (short)0, o2, false, false);
    }
  }

  // ---- epilogue: normalize, write f16 rows for the output projection ----
#pragma unroll
  for (int r = 0; r < 8; ++r) {
    const int trow = qt * 16 + rhi + r;
    const float inv = 1.0f / lrow[r];
    size_t off = ((size_t)b * cfg::T + trow) * cfg::NE + h * cfg::DH;
    ah[off + lane_col]      = (_Float16)(o0[r] * inv);
    ah[off + 16 + lane_col] = (_Float16)(o1[r] * inv);
    ah[off + 32 + lane_col] = (_Float16)(o2[r] * inv);
  }
}

// ---------------------------------------------------------------------------
// Host: orchestrate the pipeline on `stream`
// ---------------------------------------------------------------------------

extern "C" void kernel_launch(void* const* d_in, const int* in_sizes, int n_in,
                              void* d_out, int out_size, void* d_ws, size_t ws_size,
                              hipStream_t stream) {
  using namespace cfg;
  const float* x  = (const float*)d_in[0];
  const float* wq = (const float*)d_in[1];
  const float* wk = (const float*)d_in[2];
  const float* wv = (const float*)d_in[3];
  const float* wo = (const float*)d_in[4];
  const float* bo = (const float*)d_in[5];
  float* out = (float*)d_out;

  char* ws = (char*)d_ws;
  size_t off = 0;
  auto take = [&](size_t bytes) -> char* {
    char* p = ws + off;
    off = (off + bytes + 255) & ~(size_t)255;
    return p;
  };

  _Float16* xh   = (_Float16*)take((size_t)M * NE * 2);
  _Float16* wqT  = (_Float16*)take((size_t)NE * NE * 2);
  _Float16* wkT  = (_Float16*)take((size_t)(HKV * DH) * NE * 2);
  _Float16* wvT  = (_Float16*)take((size_t)(HKV * DH) * NE * 2);
  _Float16* woT  = (_Float16*)take((size_t)NE * NE * 2);
  float*    qf32 = (float*)take((size_t)M * (HQ * DH) * 4);
  float*    kf32 = (float*)take((size_t)M * (HKV * DH) * 4);
  _Float16* qh   = (_Float16*)take((size_t)B * HQ * T * DP * 2);
  _Float16* kh   = (_Float16*)take((size_t)B * HKV * T * DP * 2);
  _Float16* vt   = (_Float16*)take((size_t)B * HKV * DH * T * 2);
  _Float16* ah   = (_Float16*)take((size_t)M * NE * 2);
  (void)ws_size;

  // 1) x -> f16
  {
    size_t n = (size_t)M * NE;
    convert_f32_to_f16<<<(unsigned)((n + 255) / 256), 256, 0, stream>>>(x, xh, n);
  }
  // 2) weight transposes (K x N -> N-major f16)
  {
    size_t n = (size_t)NE * NE;
    transpose_to_f16<<<(unsigned)((n + 255) / 256), 256, 0, stream>>>(wq, wqT, NE, NE);
    transpose_to_f16<<<(unsigned)((n + 255) / 256), 256, 0, stream>>>(wo, woT, NE, NE);
    size_t nk = (size_t)NE * (HKV * DH);
    transpose_to_f16<<<(unsigned)((nk + 255) / 256), 256, 0, stream>>>(wk, wkT, NE, HKV * DH);
    transpose_to_f16<<<(unsigned)((nk + 255) / 256), 256, 0, stream>>>(wv, wvT, NE, HKV * DH);
  }
  // 3) q / k / v projections
  {
    // q: 16x64 strips -> (16384/16) * (384/64) waves
    gemm_wmma_multi<4><<<(M / 16) * (NE / 64), 32, 0, stream>>>(
        xh, NE, wqT, NE, nullptr, qf32, HQ * DH, NE / 64, NE);
    // k: 16x48 strips -> (16384/16) * (96/48) waves
    gemm_wmma_multi<3><<<(M / 16) * ((HKV * DH) / 48), 32, 0, stream>>>(
        xh, NE, wkT, NE, nullptr, kf32, HKV * DH, (HKV * DH) / 48, NE);
    // v: full 96-wide strip per wave, scattered transposed
    gemm_wmma_vproj<<<M / 16, 32, 0, stream>>>(xh, NE, wvT, NE, vt, NE);
  }
  // 4) RoPE + pack to padded f16 layouts
  {
    size_t nq = (size_t)B * T * HQ * 32;
    size_t nk = (size_t)B * T * HKV * 32;
    rope_pack<<<(unsigned)((nq + 255) / 256), 256, 0, stream>>>(qf32, qh, HQ);
    rope_pack<<<(unsigned)((nk + 255) / 256), 256, 0, stream>>>(kf32, kh, HKV);
  }
  // 5) flash attention
  {
    int blocks = B * HQ * (T / 16);  // 8192 waves
    attn_flash<<<blocks, 32, 0, stream>>>(qh, kh, vt, ah);
  }
  // 6) output projection + bias -> f32 result
  {
    gemm_wmma_multi<4><<<(M / 16) * (NE / 64), 32, 0, stream>>>(
        ah, NE, woT, NE, bo, out, NE, NE / 64, NE);
  }
}
